// BaseAttention_13795434955497
// MI455X (gfx1250) — matrix operand order repaired and checked by reading, not compiled, not run on hardware
//
#include <hip/hip_runtime.h>
#include <hip/hip_bf16.h>

// ---------------------------------------------------------------------------
// Linear attention block (B=4, L=4096, DM=1024, H=16, D=64) for gfx1250.
// Hot path: 4 GEMMs [16384x1024]@[1024x1024] on v_wmma_f32_16x16x32_bf16.
// Block tile 256x128, wave tile 64x64 (16 WMMA : 16 ds_load_b128 per K-step).
// All fp32->bf16 conversion / weight transposition hoisted out of the GEMM.
// ---------------------------------------------------------------------------

typedef __attribute__((ext_vector_type(16))) __bf16 v16bf;
typedef __attribute__((ext_vector_type(8)))  __bf16 v8bf;
typedef __attribute__((ext_vector_type(2)))  __bf16 v2bf;
typedef __attribute__((ext_vector_type(8)))  float  v8f;
typedef __attribute__((ext_vector_type(4)))  unsigned int v4u;   // 16B chunk

#define GM 16384   // B*L
#define GN 1024    // DM
#define GK 1024    // DM
#define LDA 40     // LDS row stride in bf16 elements (80 bytes, 16B aligned)

union AFrag { v16bf v; v8bf h[2]; };

// ---------------------------------------------------------------------------
// fp32 -> bf16 bulk convert (query/key/value), 8 elements/thread.
// ---------------------------------------------------------------------------
__global__ __launch_bounds__(256) void cvt_bf16_kernel(
    const float* __restrict__ s0, const float* __restrict__ s1, const float* __restrict__ s2,
    __bf16* __restrict__ d0, __bf16* __restrict__ d1, __bf16* __restrict__ d2)
{
    const float* __restrict__ s = (blockIdx.z == 0) ? s0 : (blockIdx.z == 1) ? s1 : s2;
    __bf16*      __restrict__ d = (blockIdx.z == 0) ? d0 : (blockIdx.z == 1) ? d1 : d2;
    const size_t i = ((size_t)blockIdx.x * 256 + threadIdx.x) * 8;
    const float4 a = *(const float4*)(s + i);
    const float4 b = *(const float4*)(s + i + 4);
    v8bf pk = { (__bf16)a.x, (__bf16)a.y, (__bf16)a.z, (__bf16)a.w,
                (__bf16)b.x, (__bf16)b.y, (__bf16)b.z, (__bf16)b.w };
    *(v8bf*)(d + i) = pk;
}

// ---------------------------------------------------------------------------
// Weight prep: W fp32 [K][N] -> bf16 W^T [N][K], LDS-tiled 64x64 transpose.
// blockIdx.z selects one of the 4 weight matrices.
// ---------------------------------------------------------------------------
__global__ __launch_bounds__(256) void wt_kernel(
    const float* __restrict__ W0, const float* __restrict__ W1,
    const float* __restrict__ W2, const float* __restrict__ W3,
    __bf16* __restrict__ Tbase)
{
    const float* __restrict__ W =
        (blockIdx.z == 0) ? W0 : (blockIdx.z == 1) ? W1 : (blockIdx.z == 2) ? W2 : W3;
    __bf16* __restrict__ T = Tbase + (size_t)blockIdx.z * (size_t)GK * GN;

    __shared__ alignas(16) __bf16 tile[64 * 72];   // [n][k], padded rows (144B)
    const int k0 = blockIdx.x * 64;
    const int n0 = blockIdx.y * 64;
    const int tid = threadIdx.x;

    // load 64(k) x 64(n) fp32, store transposed bf16 into LDS
    #pragma unroll
    for (int i = 0; i < 4; ++i) {
        const int c  = tid + i * 256;       // 1024 float4 chunks
        const int kk = c >> 4;              // 0..63
        const int n4 = c & 15;
        const float4 v = *(const float4*)(W + (size_t)(k0 + kk) * GN + n0 + n4 * 4);
        tile[(n4 * 4 + 0) * 72 + kk] = (__bf16)v.x;
        tile[(n4 * 4 + 1) * 72 + kk] = (__bf16)v.y;
        tile[(n4 * 4 + 2) * 72 + kk] = (__bf16)v.z;
        tile[(n4 * 4 + 3) * 72 + kk] = (__bf16)v.w;
    }
    __syncthreads();

    // write W^T rows (n-major, k contiguous) with 16B stores
    #pragma unroll
    for (int i = 0; i < 2; ++i) {
        const int c  = tid + i * 256;       // 512 v8bf chunks
        const int nn = c >> 3;              // 0..63
        const int k8 = c & 7;
        const v8bf o = *(const v8bf*)&tile[nn * 72 + k8 * 8];
        *(v8bf*)(T + (size_t)(n0 + nn) * GK + k0 + k8 * 8) = o;
    }
}

// ---------------------------------------------------------------------------
// GEMM: Out[M,N] = f32( A[M,K] @ Wt[N,K]^T ) + bias[N], A/Wt bf16 row-major.
// Block tile 256x128, BK=32, 256 threads = 8 waves (4m x 2n), wave tile 64x64
// = 4x4 WMMA tiles. Pipeline: regs hold tile t+2, LDS buffer cur^1 receives
// tile t+1, compute reads tile t from LDS buffer cur. One barrier per K-step.
// ---------------------------------------------------------------------------
__global__ __launch_bounds__(256) void gemm_bf16_wmma(
    const __bf16* __restrict__ A0, const __bf16* __restrict__ A1, const __bf16* __restrict__ A2,
    const __bf16* __restrict__ T0, const __bf16* __restrict__ T1, const __bf16* __restrict__ T2,
    const float* __restrict__ B0, const float* __restrict__ B1, const float* __restrict__ B2,
    float* __restrict__ O0, float* __restrict__ O1, float* __restrict__ O2)
{
    const __bf16* __restrict__ A  = (blockIdx.z == 0) ? A0 : (blockIdx.z == 1) ? A1 : A2;
    const __bf16* __restrict__ Wt = (blockIdx.z == 0) ? T0 : (blockIdx.z == 1) ? T1 : T2;
    const float*  __restrict__ Bi = (blockIdx.z == 0) ? B0 : (blockIdx.z == 1) ? B1 : B2;
    float*        __restrict__ O  = (blockIdx.z == 0) ? O0 : (blockIdx.z == 1) ? O1 : O2;

    const int m0 = blockIdx.y * 256;
    const int n0 = blockIdx.x * 128;

    __shared__ alignas(16) __bf16 As[2][256 * LDA];   // [m][k]  (20 KB each)
    __shared__ alignas(16) __bf16 Bs[2][128 * LDA];   // [n][k]  (10 KB each)

    const int tid  = threadIdx.x;
    const int wave = tid >> 5;
    const int lane = tid & 31;
    const int wm   = wave & 3;        // m offset wm*64
    const int wn   = wave >> 2;       // n offset wn*64
    const int half = lane >> 4;       // K sub-range select per ISA layout
    const int ml   = lane & 15;       // row (A) / col (B,C) within 16

    // Staging: A = 1024 16B-chunks (4/thread), B = 512 16B-chunks (2/thread).
    const int sr = tid >> 2;          // row base; A row = sr + 64*i, B row same
    const int sc = tid & 3;           // 16B chunk in row (k offset sc*8 elems)

    v8f acc[4][4] = {};
    v4u ra[4], rb[2];

    auto load_tile = [&](int kt) {
        #pragma unroll
        for (int i = 0; i < 4; ++i)
            ra[i] = *(const v4u*)(A + (size_t)(m0 + sr + 64 * i) * GK + kt + sc * 8);
        #pragma unroll
        for (int i = 0; i < 2; ++i)
            rb[i] = *(const v4u*)(Wt + (size_t)(n0 + sr + 64 * i) * GK + kt + sc * 8);
    };

    auto store_tile = [&](__bf16* as, __bf16* bs) {
        #pragma unroll
        for (int i = 0; i < 4; ++i)
            *(v4u*)&as[(sr + 64 * i) * LDA + sc * 8] = ra[i];
        #pragma unroll
        for (int i = 0; i < 2; ++i)
            *(v4u*)&bs[(sr + 64 * i) * LDA + sc * 8] = rb[i];
    };

    auto compute = [&](const __bf16* as, const __bf16* bs) {
        AFrag af[4];
        #pragma unroll
        for (int tm = 0; tm < 4; ++tm) {
            const int row = wm * 64 + tm * 16 + ml;
            af[tm].h[0] = *(const v8bf*)&as[row * LDA + 8 * half];       // k = 8h..8h+7
            af[tm].h[1] = *(const v8bf*)&as[row * LDA + 16 + 8 * half]; // k = 16+8h..+7
        }
        AFrag bfr[4];
        #pragma unroll
        for (int tn = 0; tn < 4; ++tn) {
            const int col = wn * 64 + tn * 16 + ml;
            bfr[tn].h[0] = *(const v8bf*)&bs[col * LDA + 8 * half];       // k = 8h..8h+7
            bfr[tn].h[1] = *(const v8bf*)&bs[col * LDA + 16 + 8 * half]; // k = 16+8h..+7
        }
        #pragma unroll
        for (int tm = 0; tm < 4; ++tm)
            #pragma unroll
            for (int tn = 0; tn < 4; ++tn)
                acc[tm][tn] = __builtin_amdgcn_wmma_f32_16x16x32_bf16(
                    false, af[tm].v, false, bfr[tn].v,
                    (short)0, acc[tm][tn], false, false);
    };

    // ---- pipelined main loop ---------------------------------------------
    const int NT = GK / 32;                 // 32 K-steps
    load_tile(0);
    store_tile(As[0], Bs[0]);
    load_tile(32);
    __syncthreads();

    #pragma unroll 2
    for (int t = 0; t < NT; ++t) {
        const int cur = t & 1;
        if (t + 1 < NT) {
            store_tile(As[cur ^ 1], Bs[cur ^ 1]);   // tile t+1 (already in regs)
            if (t + 2 < NT)
                load_tile((t + 2) * 32);            // tile t+2 -> regs
        }
        compute(As[cur], Bs[cur]);                  // tile t
        __syncthreads();
    }

    // ---- epilogue: bias add, fp32 store -----------------------------------
    #pragma unroll
    for (int tn = 0; tn < 4; ++tn) {
        const int col = n0 + wn * 64 + tn * 16 + ml;
        const float bv = Bi[col];
        #pragma unroll
        for (int tm = 0; tm < 4; ++tm) {
            #pragma unroll
            for (int r = 0; r < 8; ++r) {
                const int row = m0 + wm * 64 + tm * 16 + r + 8 * half;
                O[(size_t)row * GN + col] = acc[tm][tn][r] + bv;
            }
        }
    }
}

// ---------------------------------------------------------------------------
// Linear-attention context: one wave per (b,h,l) row of D=64.
//   fq = elu(q)+1, fk = elu(k)+1
//   ctx = fq * sum_d(fk*v) / (fq * sum_d(fk) + eps), emitted as bf16
//   (it is the A operand of the output-projection GEMM).
// ---------------------------------------------------------------------------
__global__ __launch_bounds__(256) void ctx_kernel(
    const float* __restrict__ q, const float* __restrict__ k,
    const float* __restrict__ v, __bf16* __restrict__ ctxb)
{
    const int gw   = (int)((blockIdx.x * blockDim.x + threadIdx.x) >> 5);  // row id
    const int lane = threadIdx.x & 31;
    const size_t base = (size_t)gw * 64 + (size_t)lane * 2;

    const float2 qv = *(const float2*)(q + base);
    const float2 kv = *(const float2*)(k + base);
    const float2 vv = *(const float2*)(v + base);

    const float fq0 = qv.x > 0.f ? qv.x + 1.f : __expf(qv.x);
    const float fq1 = qv.y > 0.f ? qv.y + 1.f : __expf(qv.y);
    const float fk0 = kv.x > 0.f ? kv.x + 1.f : __expf(kv.x);
    const float fk1 = kv.y > 0.f ? kv.y + 1.f : __expf(kv.y);

    float s_k  = fk0 + fk1;
    float s_kv = fk0 * vv.x + fk1 * vv.y;
    #pragma unroll
    for (int off = 16; off > 0; off >>= 1) {
        s_k  += __shfl_xor(s_k,  off, 32);
        s_kv += __shfl_xor(s_kv, off, 32);
    }

    const float o0 = fq0 * s_kv / (fq0 * s_k + 1e-6f);
    const float o1 = fq1 * s_kv / (fq1 * s_k + 1e-6f);
    v2bf o = { (__bf16)o0, (__bf16)o1 };
    *(v2bf*)(ctxb + base) = o;
}

// ---------------------------------------------------------------------------
// Residual + LayerNorm over DM=1024 (biased variance, eps=1e-3).
// One 256-thread block per row; 4 elements per thread.
// ---------------------------------------------------------------------------
__global__ __launch_bounds__(256) void ln_kernel(
    const float* __restrict__ qin, const float* __restrict__ attn,
    const float* __restrict__ gamma, const float* __restrict__ beta,
    float* __restrict__ out)
{
    __shared__ float red[16];
    const int row = blockIdx.x;
    const int t   = threadIdx.x;
    const size_t base = (size_t)row * GN + (size_t)t * 4;

    const float4 xq = *(const float4*)(qin + base);
    const float4 xa = *(const float4*)(attn + base);
    float4 x;
    x.x = xq.x + xa.x; x.y = xq.y + xa.y; x.z = xq.z + xa.z; x.w = xq.w + xa.w;

    float s  = x.x + x.y + x.z + x.w;
    float s2 = x.x * x.x + x.y * x.y + x.z * x.z + x.w * x.w;
    #pragma unroll
    for (int off = 16; off > 0; off >>= 1) {
        s  += __shfl_xor(s,  off, 32);
        s2 += __shfl_xor(s2, off, 32);
    }
    const int wv = t >> 5;
    if ((t & 31) == 0) { red[wv] = s; red[8 + wv] = s2; }
    __syncthreads();
    if (t == 0) {
        float S = 0.f, S2 = 0.f;
        #pragma unroll
        for (int i = 0; i < 8; ++i) { S += red[i]; S2 += red[8 + i]; }
        red[0] = S; red[8] = S2;
    }
    __syncthreads();

    const float mean = red[0] * (1.0f / GN);
    const float var  = red[8] * (1.0f / GN) - mean * mean;
    const float inv  = rsqrtf(var + 1e-3f);

    const float4 g = *(const float4*)(gamma + (size_t)t * 4);
    const float4 b = *(const float4*)(beta  + (size_t)t * 4);
    float4 o;
    o.x = (x.x - mean) * inv * g.x + b.x;
    o.y = (x.y - mean) * inv * g.y + b.y;
    o.z = (x.z - mean) * inv * g.z + b.z;
    o.w = (x.w - mean) * inv * g.w + b.w;
    *(float4*)(out + base) = o;
}

// ---------------------------------------------------------------------------
extern "C" void kernel_launch(void* const* d_in, const int* in_sizes, int n_in,
                              void* d_out, int out_size, void* d_ws, size_t ws_size,
                              hipStream_t stream)
{
    const float* query = (const float*)d_in[0];
    const float* key   = (const float*)d_in[1];
    const float* value = (const float*)d_in[2];
    const float* Wq = (const float*)d_in[3];  const float* bq = (const float*)d_in[4];
    const float* Wk = (const float*)d_in[5];  const float* bk = (const float*)d_in[6];
    const float* Wv = (const float*)d_in[7];  const float* bv = (const float*)d_in[8];
    const float* Wo = (const float*)d_in[9];  const float* bo = (const float*)d_in[10];
    const float* gamma = (const float*)d_in[11];
    const float* beta  = (const float*)d_in[12];
    float* out = (float*)d_out;

    const size_t NE = (size_t)GM * GN;       // 16M activations
    const size_t NW = (size_t)GK * GN;       // 1M weights

    // workspace layout (~296 MB total):
    float*  P0 = (float*)d_ws;               // q proj fp32           (64 MB)
    float*  P1 = P0 + NE;                    // k proj, then attn_out (64 MB)
    float*  P2 = P1 + NE;                    // v proj fp32           (64 MB)
    __bf16* Xb = (__bf16*)(P2 + NE);         // query bf16, then ctx  (32 MB)
    __bf16* Kb = Xb + NE;                    // key bf16              (32 MB)
    __bf16* Vb = Kb + NE;                    // value bf16            (32 MB)
    __bf16* Wt = Vb + NE;                    // 4x W^T bf16            (8 MB)

    const dim3 blk(256);

    // 0a) activations fp32 -> bf16
    cvt_bf16_kernel<<<dim3((unsigned)(NE / (256 * 8)), 1, 3), blk, 0, stream>>>(
        query, key, value, Xb, Kb, Vb);
    // 0b) weights fp32 [K][N] -> bf16 W^T [N][K]
    wt_kernel<<<dim3(GK / 64, GN / 64, 4), blk, 0, stream>>>(Wq, Wk, Wv, Wo, Wt);

    // 1) Q/K/V projections (three GEMMs via blockIdx.z)
    gemm_bf16_wmma<<<dim3(GN / 128, GM / 256, 3), blk, 0, stream>>>(
        Xb, Kb, Vb, Wt, Wt + NW, Wt + 2 * NW, bq, bk, bv, P0, P1, P2);

    // 2) linear-attention context (wave per 64-wide head row) -> bf16 in Xb
    const int rows = GM * (GN / 64);                 // B*H*L = 262144
    ctx_kernel<<<dim3(rows / 8), blk, 0, stream>>>(P0, P1, P2, Xb);

    // 3) output projection: attn_out = ctx @ Wo + bo -> P1
    gemm_bf16_wmma<<<dim3(GN / 128, GM / 256, 1), blk, 0, stream>>>(
        Xb, Xb, Xb, Wt + 3 * NW, Wt + 3 * NW, Wt + 3 * NW, bo, bo, bo, P1, P1, P1);

    // 4) residual + layernorm -> d_out
    ln_kernel<<<dim3(GM), blk, 0, stream>>>(query, P1, gamma, beta, out);
}
